// Decay2D_29094108463321
// MI455X (gfx1250) — matrix operand order repaired and checked by reading, not compiled, not run on hardware
//
#include <hip/hip_runtime.h>
#include <math.h>

// ---------------------------------------------------------------------------
// Decayed linear attention, chunked-parallel, bf16 WMMA (gfx1250 / CDNA5).
// Effective decay kernel of the reference's smear(): w(Delta) = d^ceil(Delta/2).
// Parity split: mem[t] = E[t] + d*E[t-1], E = decay-d scan over stride-2 class.
// Chunked (C=64): carry two 64x64 states (Se,So); chunk decay d^32.
// LDS tile staging uses the Tensor Data Mover (TENSOR_LOAD_TO_LDS).
// ---------------------------------------------------------------------------

#define DECAYF 0.9f
#define SCALEF 0.31622776601683794f   // sqrt(1-DECAY)
#define D32F   0.03433683820292512f   // DECAY^32

typedef __bf16 bf16_t;
typedef __attribute__((ext_vector_type(16))) __bf16 bf16x16;
typedef __attribute__((ext_vector_type(8)))  __bf16 bf16x8;
typedef __attribute__((ext_vector_type(8)))  float  v8f;
typedef __attribute__((ext_vector_type(4)))  unsigned int u32x4;
typedef __attribute__((ext_vector_type(4)))  int    i32x4;
typedef __attribute__((ext_vector_type(8)))  int    i32x8;

static __device__ __forceinline__ v8f wmma_bf16(bf16x16 a, bf16x16 b, v8f c) {
  // D = A(16x32) * B(32x16) + C, f32 accumulate
  return __builtin_amdgcn_wmma_f32_16x16x32_bf16(false, a, false, b, (short)0, c,
                                                 false, false);
}

static __device__ __forceinline__ v8f vzero8() {
  v8f z;
#pragma unroll
  for (int i = 0; i < 8; ++i) z[i] = 0.f;
  return z;
}

// Low 32 bits of a generic pointer to LDS = byte offset within the workgroup's
// LDS allocation (aperture mapping truncates addr[31:0]).
static __device__ __forceinline__ unsigned lds_off(const void* p) {
  return (unsigned)(unsigned long long)(uintptr_t)p;
}

// ---------------------------------------------------------------------------
// Tensor Data Mover: 2D bf16 tile load, Global -> LDS, one instruction.
// D# built per CDNA5 ISA 8.3/8.4:
//   group0: count=1 | lds_addr | global_addr | type=2
//   group1: data_size=1(2B), tensor_dim0/1, tile_dim0/1, dim0 stride
// Issue from a single wave; completion via s_wait_tensorcnt 0.
// ---------------------------------------------------------------------------
static __device__ __forceinline__ void tdm_load_bf16_2d(unsigned ldsoff,
                                                        const bf16_t* gptr,
                                                        int rows, int cols,
                                                        int row_stride) {
  unsigned long long ga = (unsigned long long)(uintptr_t)gptr;
  u32x4 g0;
  g0[0] = 1u;                                          // count=1, user D#
  g0[1] = ldsoff;                                      // lds_addr (bytes)
  g0[2] = (unsigned)(ga & 0xffffffffu);                // global_addr[31:0]
  g0[3] = (unsigned)((ga >> 32) & 0x01ffffffu)         // global_addr[56:32]
          | (2u << 30);                                // type=2 ("image")
  i32x8 g1;
  g1[0] = (1 << 16);                                   // data_size=1 (2 bytes)
  g1[1] = (cols & 0xffff) << 16;                       // tensor_dim0 lo16
  g1[2] = ((cols >> 16) & 0xffff)                      // tensor_dim0 hi16
          | ((rows & 0xffff) << 16);                   // tensor_dim1 lo16
  g1[3] = ((rows >> 16) & 0xffff)                      // tensor_dim1 hi16
          | ((cols & 0xffff) << 16);                   // tile_dim0
  g1[4] = (rows & 0xffff);                             // tile_dim1 (tile_dim2=0)
  g1[5] = row_stride;                                  // tensor_dim0_stride lo32
  g1[6] = 0;                                           // stride hi16 / dim1_stride lo
  g1[7] = 0;
  i32x4 gz;
  gz[0] = gz[1] = gz[2] = gz[3] = 0;                   // groups 2/3 unused (2D)
#if defined(__clang_major__) && (__clang_major__ >= 23)
  i32x8 gz8;
#pragma unroll
  for (int i = 0; i < 8; ++i) gz8[i] = 0;
  __builtin_amdgcn_tensor_load_to_lds(g0, g1, gz, gz, gz8, 0);
#else
  __builtin_amdgcn_tensor_load_to_lds(g0, g1, gz, gz, 0);
#endif
}

// A-matrix 16x32 bf16 fragment (ISA 7.12.2): lane group p=lane>>4:
// halves 0..7 -> K = kb+8p+0..7 ; halves 8..15 -> K = kb+16+8p+0..7
static __device__ __forceinline__ bf16x16 frag_a_row(const bf16_t* base, int ld,
                                                     int row, int kb, int p) {
  const bf16_t* rp = base + (size_t)row * ld + kb + 8 * p;
  bf16x8 lo = *(const bf16x8*)rp;
  bf16x8 hi = *(const bf16x8*)(rp + 16);
  bf16x16 o;
#pragma unroll
  for (int i = 0; i < 8; ++i) { o[i] = lo[i]; o[8 + i] = hi[i]; }
  return o;
}

// Same as above but converting from f32 source on the fly.
static __device__ __forceinline__ bf16x16 frag_a_row_f32(const float* base, int ld,
                                                         int row, int kb, int p) {
  const float* rp = base + (size_t)row * ld + kb + 8 * p;
  bf16x16 o;
#pragma unroll
  for (int i = 0; i < 8; ++i) {
    o[i]     = (bf16_t)rp[i];
    o[8 + i] = (bf16_t)rp[16 + i];
  }
  return o;
}

// B-matrix 32x16 bf16 fragment: column n = lane&15 is row `row` of the source;
// lane group p: halves 0..7 -> K = kb+8p+0..7 ; halves 8..15 -> K = kb+16+8p+0..7
static __device__ __forceinline__ bf16x16 frag_b_row(const bf16_t* base, int ld,
                                                     int row, int kb, int p) {
  const bf16_t* rp = base + (size_t)row * ld + kb + 8 * p;
  bf16x8 lo = *(const bf16x8*)rp;
  bf16x8 hi = *(const bf16x8*)(rp + 16);
  bf16x16 o;
#pragma unroll
  for (int i = 0; i < 8; ++i) { o[i] = lo[i]; o[8 + i] = hi[i]; }
  return o;
}

// B fragment gathered with element stride (column of a row-major matrix).
// base must already point at (k=0 of this k-step, column n).
static __device__ __forceinline__ bf16x16 frag_b_gather(const bf16_t* base,
                                                        int stride, int p) {
  bf16x16 o;
#pragma unroll
  for (int h = 0; h < 16; ++h) o[h] = base[((h & 7) + 8 * p + ((h & 8) ? 16 : 0)) * stride];
  return o;
}

// ---------------------------------------------------------------------------
// Kernel 0: convert weights to bf16 (each matrix is 65536 elements).
// ---------------------------------------------------------------------------
__global__ __launch_bounds__(256) void convert_weights(
    const float* Wk, const float* Wv, const float* Wq, const float* Wo,
    bf16_t* wk, bf16_t* wv, bf16_t* wq, bf16_t* wo) {
  int i = blockIdx.x * 256 + threadIdx.x;
  if (i < 64 * 1024) {
    wk[i] = (bf16_t)Wk[i];
    wv[i] = (bf16_t)Wv[i];
    wq[i] = (bf16_t)Wq[i];
    wo[i] = (bf16_t)Wo[i];
  }
}

// ---------------------------------------------------------------------------
// Kernel 1: fused projections. One wave computes a 16-row tile of all three
// 64-wide outputs (12 accumulator tiles), reduction K = 1024 (32 WMMA steps).
// k,q get sigmoid; results stored bf16. grid = 64, block = 256 (8 waves).
// ---------------------------------------------------------------------------
__global__ __launch_bounds__(256) void proj_kernel(
    const float* x, const bf16_t* wk, const bf16_t* wq, const bf16_t* wv,
    const float* bk, const float* bq, const float* bv,
    bf16_t* Kc, bf16_t* Qc, bf16_t* Vc) {
  const int tid = threadIdx.x;
  const int lane = tid & 31, wid = tid >> 5;
  const int p = lane >> 4, lm = lane & 15;
  const int row0 = (blockIdx.x * 8 + wid) * 16;

  const bf16_t* Ws[3] = {wk, wq, wv};
  v8f acc[3][4];
#pragma unroll
  for (int m = 0; m < 3; ++m)
#pragma unroll
    for (int ct = 0; ct < 4; ++ct) acc[m][ct] = vzero8();

  for (int ks = 0; ks < 32; ++ks) {
    const int kb = ks * 32;
    bf16x16 a = frag_a_row_f32(x, 1024, row0 + lm, kb, p);
#pragma unroll
    for (int m = 0; m < 3; ++m)
#pragma unroll
      for (int ct = 0; ct < 4; ++ct) {
        bf16x16 b = frag_b_row(Ws[m], 1024, ct * 16 + lm, kb, p);
        acc[m][ct] = wmma_bf16(a, b, acc[m][ct]);
      }
  }

  const float* Bs[3] = {bk, bq, bv};
  bf16_t* Os[3] = {Kc, Qc, Vc};
#pragma unroll
  for (int m = 0; m < 3; ++m)
#pragma unroll
    for (int ct = 0; ct < 4; ++ct) {
      const int col = ct * 16 + lm;
      const float bias = Bs[m][col];
#pragma unroll
      for (int r = 0; r < 8; ++r) {
        float val = acc[m][ct][r] + bias;
        if (m < 2) val = 1.f / (1.f + expf(-val));  // sigmoid for k,q
        Os[m][(size_t)(row0 + r + 8 * p) * 64 + col] = (bf16_t)val;
      }
    }
}

// ---------------------------------------------------------------------------
// Kernel 2a: per-chunk partial states  Pe/Po = scale * sum_{s in parity}
//            d^(31 - s/2) k_s v_s^T   (two 64x64 f32 per chunk).
// K/V tiles staged into LDS by the Tensor Data Mover.
// grid = 128 chunks, block = 256 (8 waves, 4 tiles each of 32 total).
// ---------------------------------------------------------------------------
__global__ __launch_bounds__(256) void chunk_partial(const bf16_t* Kc,
                                                     const bf16_t* Vc, float* P) {
  __shared__ bf16_t Klds[64 * 64];
  __shared__ bf16_t Vlds[64 * 64];
  __shared__ float wtab[64];
  const int g = blockIdx.x;
  const int tid = threadIdx.x;

  if (tid == 0) {  // wave 0 issues two TDM tile loads and waits on TENSORcnt
    tdm_load_bf16_2d(lds_off(Klds), Kc + (size_t)g * 4096, 64, 64, 64);
    tdm_load_bf16_2d(lds_off(Vlds), Vc + (size_t)g * 4096, 64, 64, 64);
    __builtin_amdgcn_s_wait_tensorcnt(0);
  }
  if (tid < 64) wtab[tid] = SCALEF * powf(DECAYF, (float)(31 - (tid >> 1)));
  __syncthreads();

  const int lane = tid & 31, wid = tid >> 5;
  const int p = lane >> 4, lm = lane & 15;
#pragma unroll
  for (int q4 = 0; q4 < 4; ++q4) {
    const int tt = wid * 4 + q4;
    const int par = tt >> 4;        // 0 = even-class (Pe), 1 = odd-class (Po)
    const int tile = tt & 15;
    const int tm = tile >> 2, tn = tile & 3;
    const int i_ = tm * 16 + lm, j_ = tn * 16 + lm;
    v8f acc = vzero8();
    for (int ks = 0; ks < 2; ++ks) {
      bf16x16 a;  // A[m=i, k=s] = mask(s) * w(s) * K[s][i]  (transposed gather)
#pragma unroll
      for (int h = 0; h < 16; ++h) {
        const int s = ks * 32 + ((h & 7) + 8 * p + ((h & 8) ? 16 : 0));
        const float kv =
            ((s & 1) == par) ? (float)Klds[s * 64 + i_] * wtab[s] : 0.f;
        a[h] = (bf16_t)kv;
      }
      bf16x16 b = frag_b_gather(Vlds + (ks * 32) * 64 + j_, 64, p);
      acc = wmma_bf16(a, b, acc);
    }
    float* Pb = P + ((size_t)g * 2 + par) * 4096;
#pragma unroll
    for (int r = 0; r < 8; ++r)
      Pb[(tm * 16 + r + 8 * p) * 64 + j_] = acc[r];
  }
}

// ---------------------------------------------------------------------------
// Kernel 2b: sequential prefix over the 32 chunks of each batch (elementwise):
//   Sprev[c] = state before chunk c ; S <- d^32 * S + P[c].   grid = 64.
// ---------------------------------------------------------------------------
__global__ __launch_bounds__(256) void prefix_kernel(const float* P, float* Sprev) {
  const int idx = blockIdx.x * 256 + threadIdx.x;  // 0..16383
  const int b = idx >> 12;
  const int e = idx & 4095;
  float Se = 0.f, So = 0.f;
  for (int c = 0; c < 32; ++c) {
    const size_t gg = (size_t)(b * 32 + c) * 2;
    Sprev[(gg + 0) * 4096 + e] = Se;
    Sprev[(gg + 1) * 4096 + e] = So;
    Se = D32F * Se + P[(gg + 0) * 4096 + e];
    So = D32F * So + P[(gg + 1) * 4096 + e];
  }
}

// ---------------------------------------------------------------------------
// Kernel 2c: chunk output.
//   Y = [ (QK^T ⊙ D)*scale^2 | coefSe*Q*scale | coefSo*Q*scale ] @ [V; Se; So]
// Concatenated K = 192. Q and V tiles staged by TDM.
// grid = 128 chunks, block = 256 (8 waves).
// ---------------------------------------------------------------------------
__global__ __launch_bounds__(256) void chunk_out(const bf16_t* Qc, const bf16_t* Kc,
                                                 const bf16_t* Vc,
                                                 const float* Sprev, bf16_t* Yc) {
  __shared__ bf16_t Qlds[64 * 64];
  __shared__ bf16_t Acat[64 * 192];
  __shared__ bf16_t Bcat[192 * 64];
  __shared__ float dpow[33];
  __shared__ float cS[64];
  __shared__ float cO[64];
  const int g = blockIdx.x;
  const int tid = threadIdx.x;

  if (tid == 0) {  // TDM: Q chunk -> Qlds ; V chunk -> Bcat rows 0..63
    tdm_load_bf16_2d(lds_off(Qlds), Qc + (size_t)g * 4096, 64, 64, 64);
    tdm_load_bf16_2d(lds_off(Bcat), Vc + (size_t)g * 4096, 64, 64, 64);
    __builtin_amdgcn_s_wait_tensorcnt(0);
  }
  const float* se = Sprev + ((size_t)g * 2 + 0) * 4096;
  const float* so = Sprev + ((size_t)g * 2 + 1) * 4096;
  for (int e = tid; e < 4096; e += 256) {
    Bcat[4096 + e] = (bf16_t)se[e];  // rows 64..127 = Se
    Bcat[8192 + e] = (bf16_t)so[e];  // rows 128..191 = So
  }
  if (tid < 33) dpow[tid] = powf(DECAYF, (float)tid);
  if (tid < 64) {
    const int lt = tid;
    cS[lt] = SCALEF * powf(DECAYF, (float)(lt / 2 + 1));           // same parity
    cO[lt] = SCALEF * ((lt == 0) ? DECAYF
                                 : powf(DECAYF, (float)((lt - 1) / 2 + 2)));
  }
  __syncthreads();

  // Acat columns 64..191: inter-chunk terms (parity-dependent coefficients).
  for (int idx = tid; idx < 8192; idx += 256) {
    const int lt = idx >> 7;
    const int ii = idx & 127;
    const int i_ = ii & 63;
    const float qv = (float)Qlds[lt * 64 + i_];
    float coef;
    if (ii < 64) coef = ((lt & 1) == 0) ? cS[lt] : cO[lt];  // -> Se
    else         coef = ((lt & 1) == 1) ? cS[lt] : cO[lt];  // -> So
    Acat[lt * 192 + 64 + ii] = (bf16_t)(qv * coef);
  }

  const int lane = tid & 31, wid = tid >> 5;
  const int p = lane >> 4, lm = lane & 15;

  // A' = (Q K^T) ⊙ decay mask -> Acat columns 0..63.  16 tiles / 8 waves.
#pragma unroll
  for (int q2 = 0; q2 < 2; ++q2) {
    const int tt = wid * 2 + q2;
    const int tm = tt >> 2, tn = tt & 3;
    v8f acc = vzero8();
    for (int ks = 0; ks < 2; ++ks) {
      bf16x16 a = frag_a_row(Qlds, 64, tm * 16 + lm, ks * 32, p);
      bf16x16 b = frag_b_row(Kc + (size_t)g * 4096, 64, tn * 16 + lm, ks * 32, p);
      acc = wmma_bf16(a, b, acc);
    }
#pragma unroll
    for (int r = 0; r < 8; ++r) {
      const int t = tm * 16 + r + 8 * p;
      const int s = tn * 16 + lm;
      const float val =
          (s <= t) ? acc[r] * 0.1f * dpow[((t - s) + 1) >> 1] : 0.f;  // scale^2=0.1
      Acat[t * 192 + s] = (bf16_t)val;
    }
  }
  __syncthreads();

  // Y = Acat(64x192) @ Bcat(192x64)
#pragma unroll
  for (int q2 = 0; q2 < 2; ++q2) {
    const int tt = wid * 2 + q2;
    const int tm = tt >> 2, tn = tt & 3;
    const int j_ = tn * 16 + lm;
    v8f acc = vzero8();
    for (int ks = 0; ks < 6; ++ks) {
      bf16x16 a = frag_a_row(Acat, 192, tm * 16 + lm, ks * 32, p);
      bf16x16 b = frag_b_gather(Bcat + (ks * 32) * 64 + j_, 64, p);
      acc = wmma_bf16(a, b, acc);
    }
#pragma unroll
    for (int r = 0; r < 8; ++r)
      Yc[((size_t)g * 64 + tm * 16 + r + 8 * p) * 64 + j_] = (bf16_t)acc[r];
  }
}

// ---------------------------------------------------------------------------
// Kernel 3: out = Y(8192x64) @ Wo^T(64x1024), f32 output.
// grid = (64,16), block = 256: wave = 16 rows x 64 cols, K=64.
// ---------------------------------------------------------------------------
__global__ __launch_bounds__(256) void out_gemm(const bf16_t* Yc, const bf16_t* wo,
                                                float* out) {
  const int tid = threadIdx.x;
  const int lane = tid & 31, wid = tid >> 5;
  const int p = lane >> 4, lm = lane & 15;
  const int row0 = (blockIdx.x * 8 + wid) * 16;
  const int cb = blockIdx.y * 64;
  v8f acc[4];
#pragma unroll
  for (int ct = 0; ct < 4; ++ct) acc[ct] = vzero8();
  for (int ks = 0; ks < 2; ++ks) {
    bf16x16 a = frag_a_row(Yc, 64, row0 + lm, ks * 32, p);
#pragma unroll
    for (int ct = 0; ct < 4; ++ct) {
      bf16x16 b = frag_b_row(wo, 64, cb + ct * 16 + lm, ks * 32, p);
      acc[ct] = wmma_bf16(a, b, acc[ct]);
    }
  }
#pragma unroll
  for (int ct = 0; ct < 4; ++ct)
#pragma unroll
    for (int r = 0; r < 8; ++r)
      out[(size_t)(row0 + r + 8 * p) * 1024 + cb + ct * 16 + lm] = acc[ct][r];
}

// ---------------------------------------------------------------------------
// Launch.  Inputs: x, Wk, bk, Wv, bv, Wq, bq, Wo.  Output: [4,2048,1024] f32.
// Workspace (~12.4 MB): bf16 weights, bf16 K/Q/V, f32 P & Sprev, bf16 Y.
// ---------------------------------------------------------------------------
extern "C" void kernel_launch(void* const* d_in, const int* in_sizes, int n_in,
                              void* d_out, int out_size, void* d_ws, size_t ws_size,
                              hipStream_t stream) {
  (void)in_sizes; (void)n_in; (void)out_size; (void)ws_size;
  const float* x  = (const float*)d_in[0];
  const float* Wk = (const float*)d_in[1];
  const float* bk = (const float*)d_in[2];
  const float* Wv = (const float*)d_in[3];
  const float* bv = (const float*)d_in[4];
  const float* Wq = (const float*)d_in[5];
  const float* bq = (const float*)d_in[6];
  const float* Wo = (const float*)d_in[7];

  char* ws = (char*)d_ws;
  bf16_t* wk = (bf16_t*)(ws + 0);
  bf16_t* wq = (bf16_t*)(ws + 131072);
  bf16_t* wv = (bf16_t*)(ws + 262144);
  bf16_t* wo = (bf16_t*)(ws + 393216);
  bf16_t* Kc = (bf16_t*)(ws + 524288);
  bf16_t* Qc = (bf16_t*)(ws + 524288 + 1048576);
  bf16_t* Vc = (bf16_t*)(ws + 524288 + 2 * 1048576);
  float*  P     = (float*)(ws + 3670016);
  float*  Sprev = (float*)(ws + 3670016 + 4194304);
  bf16_t* Yc    = (bf16_t*)(ws + 3670016 + 2 * 4194304);

  convert_weights<<<256, 256, 0, stream>>>(Wk, Wv, Wq, Wo, wk, wv, wq, wo);
  proj_kernel<<<64, 256, 0, stream>>>(x, wk, wq, wv, bk, bq, bv, Kc, Qc, Vc);
  chunk_partial<<<128, 256, 0, stream>>>(Kc, Vc, P);
  prefix_kernel<<<64, 256, 0, stream>>>(P, Sprev);
  chunk_out<<<128, 256, 0, stream>>>(Qc, Kc, Vc, Sprev, Yc);
  out_gemm<<<dim3(64, 16), 256, 0, stream>>>(Yc, wo, (float*)d_out);
}
